// RelationalGATLayer_2628519985725
// MI455X (gfx1250) — matrix operand order repaired and checked by reading, not compiled, not run on hardware
//
#include <hip/hip_runtime.h>

// ---------------- problem constants (match reference) ----------------
constexpr int N  = 50000;     // nodes
constexpr int E  = 800000;    // edges (before self loops)
constexpr int IN = 768;       // feature dim
constexpr int H  = 4;         // heads
constexpr int C  = 64;        // out dim per head
constexpr int HC = H * C;     // 256
constexpr int R  = 32;        // relations
constexpr float NEG_SLOPE = 0.2f;
constexpr int ET = E + N;     // edges incl. self loops

constexpr size_t pad256(size_t v) { return (v + 255) & ~(size_t)255; }

// ---------------- workspace layout (bytes, 256-aligned regions) ------
constexpr size_t OFF_H     = 0;                                        // _Float16 [N*HC]
constexpr size_t OFF_WT    = OFF_H     + pad256((size_t)N * HC * 2);   // _Float16 [HC*IN] (transposed W)
constexpr size_t OFF_ASRC  = OFF_WT    + pad256((size_t)HC * IN * 2);  // float [N*H]
constexpr size_t OFF_ADST  = OFF_ASRC  + pad256((size_t)N * H * 4);    // float [N*H]
constexpr size_t OFF_DEN   = OFF_ADST  + pad256((size_t)N * H * 4);    // float [N*H]
constexpr size_t OFF_MENC  = OFF_DEN   + pad256((size_t)N * H * 4);    // int   [N*H]
constexpr size_t OFF_P     = OFF_MENC  + pad256((size_t)N * H * 4);    // float [ET*H]
constexpr size_t OFF_S32   = OFF_P     + pad256((size_t)ET * H * 4);   // int [ET]
constexpr size_t OFF_D32   = OFF_S32   + pad256((size_t)ET * 4);       // int [ET]
constexpr size_t OFF_TIX   = OFF_D32   + pad256((size_t)ET * 4);       // int [ET]
constexpr size_t OFF_AE    = OFF_TIX   + pad256((size_t)ET * 4);       // float [(R+1)*H]
constexpr size_t OFF_CNT   = OFF_AE    + 1024;                         // int [R]

// ---------------- WMMA vector types ----------------------------------
typedef _Float16 v16h __attribute__((ext_vector_type(16)));
typedef _Float16 v8h  __attribute__((ext_vector_type(8)));
typedef float    v8f  __attribute__((ext_vector_type(8)));

union AFrag { v16h v; v8h h[2]; };

// ============== 1. init: out = bias, m = -inf enc, denom = 0, cnt = 0
__global__ __launch_bounds__(256) void k_init(float* __restrict__ out,
                                              const float* __restrict__ bias,
                                              int* __restrict__ menc,
                                              float* __restrict__ denom,
                                              int* __restrict__ cnt) {
  int i = blockIdx.x * 256 + threadIdx.x;
  if (i < N * C) out[i] = bias[i & (C - 1)];
  if (i < N * H) { menc[i] = (int)0x80000000; denom[i] = 0.f; }
  if (i < R) cnt[i] = 0;
}

// ============== 2. histogram of edge types (LDS-local then global) ===
__global__ __launch_bounds__(256) void k_hist(const long long* __restrict__ et,
                                              int* __restrict__ cnt) {
  __shared__ int sh[R];
  if (threadIdx.x < R) sh[threadIdx.x] = 0;
  __syncthreads();
  int e = blockIdx.x * 256 + threadIdx.x;
  if (e < E) atomicAdd(&sh[(int)et[e]], 1);
  __syncthreads();
  if (threadIdx.x < R) atomicAdd(&cnt[threadIdx.x], sh[threadIdx.x]);
}

// ============== 3. edge-record prep: int64 -> int32, self loops ======
__global__ __launch_bounds__(256) void k_prep(const long long* __restrict__ ei,
                                              const long long* __restrict__ et,
                                              int* __restrict__ s32,
                                              int* __restrict__ d32,
                                              int* __restrict__ tix) {
  int e = blockIdx.x * 256 + threadIdx.x;
  if (e >= ET) return;
  if (e < E) {
    s32[e] = (int)ei[e];
    d32[e] = (int)ei[E + e];
    tix[e] = (int)et[e];
  } else {               // self loop: mean edge-attr row index R
    s32[e] = d32[e] = e - E;
    tix[e] = R;
  }
}

// ============== 4. per-relation attention logits a_e_rel[r][h] =======
__global__ __launch_bounds__(256) void k_rel(const float* __restrict__ rel,
                                             const float* __restrict__ We,
                                             const float* __restrict__ att_e,
                                             float* __restrict__ a_e) {
  __shared__ float red[256];
  const int r = blockIdx.x;          // relation
  const int t = threadIdx.x;         // output column 0..255
  const float* rrow = rel + (size_t)r * IN;
  float acc = 0.f;
  for (int k = 0; k < IN; ++k) acc += rrow[k] * We[(size_t)k * HC + t];
  red[t] = acc * att_e[t];           // att_e laid out [H*C] == flat 256
  __syncthreads();
  const int cc = t & (C - 1);
  for (int s = C / 2; s > 0; s >>= 1) {
    if (cc < s) red[t] += red[t + s];
    __syncthreads();
  }
  if (cc == 0) a_e[r * H + (t >> 6)] = red[t];
}

// ============== 5. mean-relation row (linearity of the mean) =========
__global__ void k_rel_mean(const int* __restrict__ cnt, float* __restrict__ a_e) {
  int hd = threadIdx.x;
  if (hd < H) {
    float s = 0.f;
    for (int r = 0; r < R; ++r) s += (float)cnt[r] * a_e[r * H + hd];
    a_e[R * H + hd] = s / (float)E;
  }
}

// ============== 6. W -> f16 transposed, fragment-ready layout ========
// Wt[c][k]: lane fragment = 16 contiguous halves -> two b128 loads
__global__ __launch_bounds__(256) void k_wcvt(const float* __restrict__ W,
                                              _Float16* __restrict__ Wt) {
  int k = blockIdx.x * 256 + threadIdx.x;   // 0..IN-1 (grid.x = IN/256)
  int c = blockIdx.y;                       // 0..HC-1
  Wt[(size_t)c * IN + k] = (_Float16)W[(size_t)k * HC + c];
}

// ============== 7. WMMA GEMM: h16 = f16(x @ W), f32 accumulate =======
// 128 threads (4 waves); 16 rows x 256 cols per block; 4 col-tiles/wave
__global__ __launch_bounds__(128) void k_gemm(const float* __restrict__ x,
                                              const _Float16* __restrict__ Wt,
                                              _Float16* __restrict__ h16) {
  __shared__ alignas(16) _Float16 As[16 * 32];   // [row][k]

  const int t    = threadIdx.x;
  const int lane = t & 31;
  const int wave = t >> 5;            // 0..3
  const int g    = (lane >> 4) & 1;   // lane group
  const int ln   = lane & 15;
  const int rowbase = blockIdx.x * 16;

  int cols[4];
#pragma unroll
  for (int j = 0; j < 4; ++j) cols[j] = (wave * 4 + j) * 16 + ln;

  v8f acc[4] = {};

  for (int kk = 0; kk < IN; kk += 32) {
    __syncthreads();
    // stage A: 16x32 f32 -> f16, 4 elems / thread, one packed b64 store
    {
      int idx = t * 4;
      int row = idx >> 5, k = idx & 31;
      const float4 xv = *(const float4*)(x + (size_t)(rowbase + row) * IN + kk + k);
      union { _Float16 h[4]; uint2 u; } pk;
      pk.h[0] = (_Float16)xv.x; pk.h[1] = (_Float16)xv.y;
      pk.h[2] = (_Float16)xv.z; pk.h[3] = (_Float16)xv.w;
      *(uint2*)&As[row * 32 + k] = pk.u;
    }
    __syncthreads();

    // A fragment: lane(ln,g) row M=ln, K = 16g..16g+15
    AFrag a;
    a.h[0] = *(const v8h*)&As[ln * 32 + 16 * g];
    a.h[1] = *(const v8h*)&As[ln * 32 + 16 * g + 8];

    // B fragments straight from L2-resident transposed f16 W
#pragma unroll
    for (int j = 0; j < 4; ++j) {
      const _Float16* bp = Wt + (size_t)cols[j] * IN + kk + 16 * g;
      AFrag b;
      b.h[0] = *(const v8h*)bp;
      b.h[1] = *(const v8h*)(bp + 8);
      acc[j] = __builtin_amdgcn_wmma_f32_16x16x32_f16(false, a.v, false, b.v,
                                                      (short)0, acc[j], false, false);
    }
  }

  // epilogue: c[r] -> row = r + 8g
  const int row0 = rowbase + 8 * g;
#pragma unroll
  for (int j = 0; j < 4; ++j)
#pragma unroll
    for (int r = 0; r < 8; ++r)
      h16[(size_t)(row0 + r) * HC + cols[j]] = (_Float16)acc[j][r];
}

// ============== 8. node logits: a_src / a_dst  [N,H] =================
__global__ __launch_bounds__(256) void k_node(const _Float16* __restrict__ h16,
                                              const float* __restrict__ att_s,
                                              const float* __restrict__ att_d,
                                              float* __restrict__ a_src,
                                              float* __restrict__ a_dst) {
  int i = blockIdx.x * 256 + threadIdx.x;
  if (i >= N * H) return;
  int n = i >> 2, hd = i & 3;
  const v8h* hv = (const v8h*)(h16 + (size_t)n * HC + hd * C);
  const float* as = att_s + hd * C;
  const float* ad = att_d + hd * C;
  float s0 = 0.f, s1 = 0.f;
#pragma unroll
  for (int j = 0; j < 8; ++j) {
    v8h hh = hv[j];
#pragma unroll
    for (int q = 0; q < 8; ++q) {
      float f = (float)hh[q];
      s0 += f * as[j * 8 + q];
      s1 += f * ad[j * 8 + q];
    }
  }
  a_src[i] = s0;
  a_dst[i] = s1;
}

// ---------------- order-preserving float<->int encoding --------------
__device__ __forceinline__ int   enc_f(float v) { int i = __float_as_int(v); return i >= 0 ? i : (i ^ 0x7fffffff); }
__device__ __forceinline__ float dec_f(int i)   { return __int_as_float(i >= 0 ? i : (i ^ 0x7fffffff)); }

__device__ __forceinline__ float edge_alpha(int e, int hd,
                                            const int* __restrict__ s32,
                                            const int* __restrict__ d32,
                                            const int* __restrict__ tix,
                                            const float* __restrict__ a_src,
                                            const float* __restrict__ a_dst,
                                            const float* __restrict__ a_e,
                                            int& dnode) {
  int s = s32[e], d = d32[e];
  dnode = d;
  float a = a_src[s * H + hd] + a_dst[d * H + hd] + a_e[tix[e] * H + hd];
  return a > 0.f ? a : NEG_SLOPE * a;
}

// ============== 9. segment max =======================================
__global__ __launch_bounds__(256) void k_max(const int* __restrict__ s32,
                                             const int* __restrict__ d32,
                                             const int* __restrict__ tix,
                                             const float* __restrict__ a_src,
                                             const float* __restrict__ a_dst,
                                             const float* __restrict__ a_e,
                                             int* __restrict__ menc) {
  int i = blockIdx.x * 256 + threadIdx.x;
  if (i >= ET * H) return;
  int e = i >> 2, hd = i & 3, d;
  float a = edge_alpha(e, hd, s32, d32, tix, a_src, a_dst, a_e, d);
  atomicMax(&menc[d * H + hd], enc_f(a));
}

// ============== 10. exp + segment sum, cache p per edge ==============
__global__ __launch_bounds__(256) void k_exp(const int* __restrict__ s32,
                                             const int* __restrict__ d32,
                                             const int* __restrict__ tix,
                                             const float* __restrict__ a_src,
                                             const float* __restrict__ a_dst,
                                             const float* __restrict__ a_e,
                                             const int* __restrict__ menc,
                                             float* __restrict__ p,
                                             float* __restrict__ denom) {
  int i = blockIdx.x * 256 + threadIdx.x;
  if (i >= ET * H) return;
  int e = i >> 2, hd = i & 3, d;
  float a = edge_alpha(e, hd, s32, d32, tix, a_src, a_dst, a_e, d);
  float pv = __expf(a - dec_f(menc[d * H + hd]));
  p[i] = pv;
  atomicAdd(&denom[d * H + hd], pv);
}

// ============== 11. weighted scatter-aggregate (1/H folded in) =======
__global__ __launch_bounds__(256) void k_agg(const int* __restrict__ s32,
                                             const int* __restrict__ d32,
                                             const _Float16* __restrict__ h16,
                                             const float* __restrict__ p,
                                             const float* __restrict__ denom,
                                             float* __restrict__ out) {
  int i = blockIdx.x * 256 + threadIdx.x;
  if (i >= ET * C) return;
  int e = i >> 6, c = i & (C - 1);
  int s = s32[e], d = d32[e];
  const _Float16* hrow = h16 + (size_t)s * HC + c;
  float acc = 0.f;
#pragma unroll
  for (int hd = 0; hd < H; ++hd) {
    float w = p[e * H + hd] / (denom[d * H + hd] + 1e-16f);
    acc += w * (float)hrow[hd * C];
  }
  atomicAdd(&out[(size_t)d * C + c], acc * (1.0f / H));
}

// =====================================================================
extern "C" void kernel_launch(void* const* d_in, const int* in_sizes, int n_in,
                              void* d_out, int out_size, void* d_ws, size_t ws_size,
                              hipStream_t stream) {
  const float*     x     = (const float*)d_in[0];
  const long long* ei    = (const long long*)d_in[1];
  const long long* et    = (const long long*)d_in[2];
  const float*     rel   = (const float*)d_in[3];
  const float*     W     = (const float*)d_in[4];
  const float*     We    = (const float*)d_in[5];
  const float*     att_s = (const float*)d_in[6];
  const float*     att_d = (const float*)d_in[7];
  const float*     att_e = (const float*)d_in[8];
  const float*     bias  = (const float*)d_in[9];
  float* out = (float*)d_out;

  char* ws = (char*)d_ws;
  _Float16* h16   = (_Float16*)(ws + OFF_H);
  _Float16* Wt    = (_Float16*)(ws + OFF_WT);
  float*    a_src = (float*)(ws + OFF_ASRC);
  float*    a_dst = (float*)(ws + OFF_ADST);
  float*    denom = (float*)(ws + OFF_DEN);
  int*      menc  = (int*)(ws + OFF_MENC);
  float*    p     = (float*)(ws + OFF_P);
  int*      s32   = (int*)(ws + OFF_S32);
  int*      d32   = (int*)(ws + OFF_D32);
  int*      tix   = (int*)(ws + OFF_TIX);
  float*    a_e   = (float*)(ws + OFF_AE);
  int*      cnt   = (int*)(ws + OFF_CNT);

  k_init<<<(N * C + 255) / 256, 256, 0, stream>>>(out, bias, menc, denom, cnt);
  k_hist<<<(E + 255) / 256, 256, 0, stream>>>(et, cnt);
  k_prep<<<(ET + 255) / 256, 256, 0, stream>>>(ei, et, s32, d32, tix);
  k_rel<<<R, 256, 0, stream>>>(rel, We, att_e, a_e);
  k_rel_mean<<<1, 32, 0, stream>>>(cnt, a_e);
  k_wcvt<<<dim3(IN / 256, HC), 256, 0, stream>>>(W, Wt);
  k_gemm<<<N / 16, 128, 0, stream>>>(x, Wt, h16);
  k_node<<<(N * H + 255) / 256, 256, 0, stream>>>(h16, att_s, att_d, a_src, a_dst);
  k_max<<<(ET * H + 255) / 256, 256, 0, stream>>>(s32, d32, tix, a_src, a_dst, a_e, menc);
  k_exp<<<(ET * H + 255) / 256, 256, 0, stream>>>(s32, d32, tix, a_src, a_dst, a_e, menc, p, denom);
  k_agg<<<(ET * C + 255) / 256, 256, 0, stream>>>(s32, d32, h16, p, denom, out);
}
